// GATsmall_12043088298518
// MI455X (gfx1250) — matrix operand order repaired and checked by reading, not compiled, not run on hardware
//
#include <hip/hip_runtime.h>
#include <hip/hip_bf16.h>
#include <stdint.h>

typedef __attribute__((ext_vector_type(16))) _Float16 v16h;
typedef __attribute__((ext_vector_type(8)))  _Float16 v8h;
typedef __attribute__((ext_vector_type(8)))  float    v8f;

#define NN   50000
#define HH   8
#define EE   500000
#define ETOT 550000
#define NEG_SLOPE 0.2f
#define EPSA 1e-16f

static __device__ __forceinline__ float lrelu(float v) { return v > 0.f ? v : NEG_SLOPE * v; }

// order-preserving float<->uint mapping so float-max can use integer atomicMax
static __device__ __forceinline__ unsigned enc_ord(float f) {
    unsigned u = __float_as_uint(f);
    return (u & 0x80000000u) ? ~u : (u | 0x80000000u);
}
static __device__ __forceinline__ float dec_ord(unsigned u) {
    u = (u & 0x80000000u) ? (u & 0x7FFFFFFFu) : ~u;
    return __uint_as_float(u);
}

// ---------------- format conversion / packing ----------------

__global__ void f32_to_f16_kernel(const float* __restrict__ in, _Float16* __restrict__ out, int total) {
    int t = blockIdx.x * blockDim.x + threadIdx.x;
    if (t < total) out[t] = (_Float16)in[t];
}

// W [K x Cin] row-major f32  ->  Bpack [Cpad x K] f16 (transposed, zero-padded cols)
__global__ void pack_bT_kernel(const float* __restrict__ W, _Float16* __restrict__ out,
                               int K, int Cin, int Cpad) {
    int t = blockIdx.x * blockDim.x + threadIdx.x;
    if (t >= Cpad * K) return;
    int c = t / K, k = t - c * K;
    out[t] = (c < Cin) ? (_Float16)W[(size_t)k * Cin + c] : (_Float16)0.f;
}

// acc + bias, relu, pack to f16 (input for the next WMMA GEMM)
__global__ void bias_relu_pack_kernel(const float* __restrict__ acc, const float* __restrict__ bias,
                                      _Float16* __restrict__ out, int total, int cmask) {
    int t = blockIdx.x * blockDim.x + threadIdx.x;
    if (t >= total) return;
    float v = acc[t] + bias[t & cmask];
    out[t] = (_Float16)(v > 0.f ? v : 0.f);
}

// ---------------- WMMA GEMM: C[n x Cstore] = A[n x K] * B[K x Cpad] ----------------
// A: f16 row-major. Bp: f16, pre-transposed [Cpad x K]. One wave per 16x16 tile,
// 8 waves/block share a 16-row A tile staged in LDS. n % 16 == 0, K % 32 == 0.

__global__ __launch_bounds__(256) void gemm_wmma_kernel(
    const _Float16* __restrict__ A, const _Float16* __restrict__ Bp,
    float* __restrict__ Cmat, int K, int Cpad, int Cstore, int ldc) {

    __shared__ __align__(16) _Float16 lds_a[16 * 256];

    const int tid = threadIdx.x;
    {   // cooperative stage of A tile: rows [blockIdx.x*16, +16) x K halves (contiguous)
        const uint2* src = reinterpret_cast<const uint2*>(A + (size_t)blockIdx.x * 16 * K);
        uint2* dst = reinterpret_cast<uint2*>(lds_a);
        const int n4 = (16 * K) >> 2;
        for (int i = tid; i < n4; i += 256) dst[i] = src[i];
    }
    __syncthreads();

    const int wave = tid >> 5;
    const int lane = tid & 31;
    const int colTile = blockIdx.y * 8 + wave;
    if (colTile * 16 >= Cpad) return;   // wave-uniform

    const int r  = lane & 15;   // A row in tile / B col in tile
    const int hi = lane >> 4;   // upper-half-lane selector

    const _Float16* abase = lds_a + r * K + hi * 16;
    const _Float16* bbase = Bp + (size_t)(colTile * 16 + r) * K + hi * 16;

    v8f c = {};
    for (int k0 = 0; k0 < K; k0 += 32) {
        // A fragment (contiguous k, matching B): lanes<16 hold K k0..k0+15,
        // lanes>=16 hold k0+16..k0+31 -> two 16B LDS reads.
        v8h alo = *reinterpret_cast<const v8h*>(abase + k0);
        v8h ahi = *reinterpret_cast<const v8h*>(abase + k0 + 8);
        v16h a;
#pragma unroll
        for (int i = 0; i < 8; ++i) { a[i] = alo[i]; a[i + 8] = ahi[i]; }
        // B fragment: lane holds col (lane&15), 16 consecutive k (lo lanes k0.., hi lanes k0+16..)
        v16h b = *reinterpret_cast<const v16h*>(bbase + k0);
        c = __builtin_amdgcn_wmma_f32_16x16x32_f16(false, a, false, b, (short)0, c, false, false);
    }

    const int col = colTile * 16 + r;
    if (col < Cstore) {
        const size_t rowBase = (size_t)blockIdx.x * 16 + hi * 8;
#pragma unroll
        for (int v = 0; v < 8; ++v)
            Cmat[(rowBase + v) * (size_t)ldc + col] = c[v];
    }
}

// ---------------- attention coefficients: out[n,h] = sum_d h[n,h,d]*att[h,d] ----------------

__global__ void att_kernel(const float* __restrict__ h, const float* __restrict__ att,
                           float* __restrict__ out, int n, int D) {
    int t = blockIdx.x * blockDim.x + threadIdx.x;
    if (t >= n * HH) return;
    int node = t / HH, hh = t - node * HH;
    const float* hp = h + (size_t)node * HH * D + hh * D;
    const float* ap = att + hh * D;
    float s = 0.f;
    for (int d = 0; d < D; ++d) s += hp[d] * ap[d];
    out[t] = s;
}

// ---------------- edge passes (src/dst with implicit self loops for e >= EE) ----------------

static __device__ __forceinline__ void edge_sd(const int* ei, int e, int& s, int& d) {
    if (e < EE) { s = ei[e]; d = ei[EE + e]; }
    else        { s = e - EE; d = s; }
}

__global__ void edge_max_kernel(const int* __restrict__ ei, const float* __restrict__ asrc,
                                const float* __restrict__ adst, unsigned* __restrict__ m) {
    int e = blockIdx.x * blockDim.x + threadIdx.x;
    if (e >= ETOT) return;
    int s, d; edge_sd(ei, e, s, d);
#pragma unroll
    for (int h = 0; h < HH; ++h) {
        float v = lrelu(asrc[s * HH + h] + adst[d * HH + h]);
        atomicMax(&m[d * HH + h], enc_ord(v));
    }
}

__global__ void edge_expsum_kernel(const int* __restrict__ ei, const float* __restrict__ asrc,
                                   const float* __restrict__ adst, const unsigned* __restrict__ m,
                                   float* __restrict__ ssum, float* __restrict__ ex) {
    int e = blockIdx.x * blockDim.x + threadIdx.x;
    if (e >= ETOT) return;
    int s, d; edge_sd(ei, e, s, d);
#pragma unroll
    for (int h = 0; h < HH; ++h) {
        float v  = lrelu(asrc[s * HH + h] + adst[d * HH + h]);
        float xv = expf(v - dec_ord(m[d * HH + h]));
        ex[(size_t)e * HH + h] = xv;
        atomicAdd(&ssum[d * HH + h], xv);
    }
}

// one wave per edge; lane covers channels lane, lane+32, ...
__global__ void edge_agg_kernel(const int* __restrict__ ei, const float* __restrict__ hsrc,
                                const float* __restrict__ ex, const float* __restrict__ ssum,
                                float* __restrict__ acc, int C, int D) {
    int gw   = (blockIdx.x * 256 + threadIdx.x) >> 5;
    int lane = threadIdx.x & 31;
    if (gw >= ETOT) return;
    int s, d; edge_sd(ei, gw, s, d);
    int nIter = C >> 5;
    for (int i = 0; i < nIter; ++i) {
        int c = lane + (i << 5);
        int h = c / D;
        float alpha = ex[(size_t)gw * HH + h] / (ssum[d * HH + h] + EPSA);
        atomicAdd(&acc[(size_t)d * C + c], hsrc[(size_t)s * C + c] * alpha);
    }
}

// ---------------- GCN head ----------------

__global__ void edge_deg_kernel(const int* __restrict__ ei, float* __restrict__ deg) {
    int e = blockIdx.x * blockDim.x + threadIdx.x;
    if (e >= ETOT) return;
    int s, d; edge_sd(ei, e, s, d); (void)s;
    atomicAdd(&deg[d], 1.0f);
}

__global__ void dinv_kernel(const float* __restrict__ deg, float* __restrict__ dinv) {
    int t = blockIdx.x * blockDim.x + threadIdx.x;
    if (t >= NN) return;
    float g = deg[t];
    dinv[t] = g > 0.f ? rsqrtf(g) : 0.f;
}

__global__ void init_out_kernel(float* __restrict__ out, const float* __restrict__ bg, int total) {
    int t = blockIdx.x * blockDim.x + threadIdx.x;
    if (t < total) out[t] = bg[t & 7];
}

__global__ void edge_gcn_kernel(const int* __restrict__ ei, const float* __restrict__ h3,
                                const float* __restrict__ dinv, float* __restrict__ out) {
    int e = blockIdx.x * blockDim.x + threadIdx.x;
    if (e >= ETOT) return;
    int s, d; edge_sd(ei, e, s, d);
    float norm = dinv[s] * dinv[d];
#pragma unroll
    for (int j = 0; j < 8; ++j)
        atomicAdd(&out[d * 8 + j], h3[s * 8 + j] * norm);
}

// ---------------- host orchestration ----------------

extern "C" void kernel_launch(void* const* d_in, const int* in_sizes, int n_in,
                              void* d_out, int out_size, void* d_ws, size_t ws_size,
                              hipStream_t stream) {
    const float* x   = (const float*)d_in[0];
    const int*   ei  = (const int*)  d_in[1];
    const float* W1  = (const float*)d_in[2];
    const float* as1 = (const float*)d_in[3];
    const float* ad1 = (const float*)d_in[4];
    const float* b1  = (const float*)d_in[5];
    const float* W2  = (const float*)d_in[6];
    const float* as2 = (const float*)d_in[7];
    const float* ad2 = (const float*)d_in[8];
    const float* b2  = (const float*)d_in[9];
    const float* Wg  = (const float*)d_in[10];
    const float* bg  = (const float*)d_in[11];

    const int C1 = 256, C2 = 128, F = 128;

    // workspace carving (256B aligned), big buffers reused across layers
    char* ws = (char*)d_ws;
    size_t off = 0;
    auto carve = [&](size_t bytes) -> char* {
        char* p = ws + off;
        off = (off + bytes + 255) & ~(size_t)255;
        return p;
    };
    _Float16* h16   = (_Float16*)carve((size_t)NN * 256 * 2);   // x_h -> h1relu_h -> h2relu_h
    float*    gemm  = (float*)   carve((size_t)NN * 256 * 4);   // h1 -> h2
    float*    acc   = (float*)   carve((size_t)NN * 256 * 4);   // acc1 -> acc2
    float*    ex    = (float*)   carve((size_t)ETOT * HH * 4);
    float*    asrcb = (float*)   carve((size_t)NN * HH * 4);
    float*    adstb = (float*)   carve((size_t)NN * HH * 4);
    unsigned* mbuf  = (unsigned*)carve((size_t)NN * HH * 4);
    float*    ssum  = (float*)   carve((size_t)NN * HH * 4);
    _Float16* wpack = (_Float16*)carve((size_t)256 * 256 * 2);
    float*    h3    = (float*)   carve((size_t)NN * 8 * 4);
    float*    deg   = (float*)   carve((size_t)NN * 4);
    float*    dinv  = (float*)   carve((size_t)NN * 4);
    (void)ws_size; (void)in_sizes; (void)n_in;

    const int TB = 256;
    dim3 blk(TB);
    auto cdiv = [](int a, int b) { return (a + b - 1) / b; };
    const int eBlocks = cdiv(ETOT, TB);
    const int wBlocks = cdiv(ETOT, 8);   // one wave per edge, 8 waves/block

    // ---- Layer 1 GAT ----
    f32_to_f16_kernel<<<cdiv(NN * F, TB), blk, 0, stream>>>(x, h16, NN * F);
    pack_bT_kernel<<<cdiv(C1 * F, TB), blk, 0, stream>>>(W1, wpack, F, C1, C1);
    gemm_wmma_kernel<<<dim3(NN / 16, 2), blk, 0, stream>>>(h16, wpack, gemm, F, C1, C1, C1);

    att_kernel<<<cdiv(NN * HH, TB), blk, 0, stream>>>(gemm, as1, asrcb, NN, 32);
    att_kernel<<<cdiv(NN * HH, TB), blk, 0, stream>>>(gemm, ad1, adstb, NN, 32);

    hipMemsetAsync(mbuf, 0, (size_t)NN * HH * 4, stream);
    hipMemsetAsync(ssum, 0, (size_t)NN * HH * 4, stream);
    edge_max_kernel<<<eBlocks, blk, 0, stream>>>(ei, asrcb, adstb, mbuf);
    edge_expsum_kernel<<<eBlocks, blk, 0, stream>>>(ei, asrcb, adstb, mbuf, ssum, ex);
    hipMemsetAsync(acc, 0, (size_t)NN * C1 * 4, stream);
    edge_agg_kernel<<<wBlocks, blk, 0, stream>>>(ei, gemm, ex, ssum, acc, C1, 32);
    bias_relu_pack_kernel<<<cdiv(NN * C1, TB), blk, 0, stream>>>(acc, b1, h16, NN * C1, C1 - 1);

    // ---- Layer 2 GAT ----
    pack_bT_kernel<<<cdiv(C2 * C1, TB), blk, 0, stream>>>(W2, wpack, C1, C2, C2);
    gemm_wmma_kernel<<<dim3(NN / 16, 1), blk, 0, stream>>>(h16, wpack, gemm, C1, C2, C2, C2);

    att_kernel<<<cdiv(NN * HH, TB), blk, 0, stream>>>(gemm, as2, asrcb, NN, 16);
    att_kernel<<<cdiv(NN * HH, TB), blk, 0, stream>>>(gemm, ad2, adstb, NN, 16);

    hipMemsetAsync(mbuf, 0, (size_t)NN * HH * 4, stream);
    hipMemsetAsync(ssum, 0, (size_t)NN * HH * 4, stream);
    edge_max_kernel<<<eBlocks, blk, 0, stream>>>(ei, asrcb, adstb, mbuf);
    edge_expsum_kernel<<<eBlocks, blk, 0, stream>>>(ei, asrcb, adstb, mbuf, ssum, ex);
    hipMemsetAsync(acc, 0, (size_t)NN * C2 * 4, stream);
    edge_agg_kernel<<<wBlocks, blk, 0, stream>>>(ei, gemm, ex, ssum, acc, C2, 16);
    bias_relu_pack_kernel<<<cdiv(NN * C2, TB), blk, 0, stream>>>(acc, b2, h16, NN * C2, C2 - 1);

    // ---- GCN head (Wg padded 8 -> 16 cols, same WMMA path) ----
    pack_bT_kernel<<<cdiv(16 * C2, TB), blk, 0, stream>>>(Wg, wpack, C2, 8, 16);
    gemm_wmma_kernel<<<dim3(NN / 16, 1), blk, 0, stream>>>(h16, wpack, h3, C2, 16, 8, 8);

    hipMemsetAsync(deg, 0, (size_t)NN * 4, stream);
    edge_deg_kernel<<<eBlocks, blk, 0, stream>>>(ei, deg);
    dinv_kernel<<<cdiv(NN, TB), blk, 0, stream>>>(deg, dinv);
    init_out_kernel<<<cdiv(out_size, TB), blk, 0, stream>>>((float*)d_out, bg, out_size);
    edge_gcn_kernel<<<eBlocks, blk, 0, stream>>>(ei, h3, dinv, (float*)d_out);
}
